// OnlySupLoss_13683765805850
// MI455X (gfx1250) — matrix operand order repaired and checked by reading, not compiled, not run on hardware
//
#include <hip/hip_runtime.h>
#include <hip/hip_bf16.h>

#define B_   128
#define S_   512
#define D_   768
#define S2_  510
#define SEP_ 103
#define MROWS_ 65280   // B_*S2_

typedef __attribute__((ext_vector_type(16))) __bf16 v16bf;
typedef __attribute__((ext_vector_type(8)))  float  v8f;

union BF16x16 { uint4 u4[2]; v16bf v; };
static_assert(sizeof(BF16x16) == 32, "frag size");

__device__ __forceinline__ unsigned f2bf_pk(float a, float b) {
    unsigned ua = __float_as_uint(a);
    unsigned ub = __float_as_uint(b);
    ua = (ua + 0x7FFFu + ((ua >> 16) & 1u)) >> 16;   // RNE to bf16
    ub = (ub + 0x7FFFu + ((ub >> 16) & 1u)) >> 16;
    return (ua & 0xFFFFu) | (ub << 16);
}
__device__ __forceinline__ unsigned short f2bf1(float a) {
    unsigned ua = __float_as_uint(a);
    return (unsigned short)((ua + 0x7FFFu + ((ua >> 16) & 1u)) >> 16);
}

// ---------------- sep position per row (first SEP_ID) --------------------
__global__ __launch_bounds__(256) void k_sep(const int* __restrict__ ids,
                                             int* __restrict__ sep) {
    __shared__ int smin;
    if (threadIdx.x == 0) smin = S_;
    __syncthreads();
    int b = blockIdx.x;
    int best = S_;
    for (int s = threadIdx.x; s < S_; s += 256)
        if (ids[b * S_ + s] == SEP_) best = min(best, s);
    atomicMin(&smin, best);
    __syncthreads();
    if (threadIdx.x == 0) sep[b] = smin;
}

// --------- pack fp32 weights into bf16 WMMA B-fragment order -------------
// dst dword id = (((t*kchunks + c)*32 + lane)*8 + v)
// B layout (32x16, KxN, bf16): lane -> n = lane&15, h = lane>>4;
//   vgpr v holds K pair k = c*32 + h*16 + v*2 + {0,1}
__global__ __launch_bounds__(256) void k_pack_w(const float* __restrict__ W0,
                                                const float* __restrict__ W1,
                                                unsigned* __restrict__ dst,
                                                int kchunks, int K,
                                                int tilesPerPool, int total) {
    int id = blockIdx.x * 256 + threadIdx.x;
    if (id >= total) return;
    int v    = id & 7;
    int lane = (id >> 3) & 31;
    int c    = (id >> 8) % kchunks;
    int t    = (id >> 8) / kchunks;
    const float* W = (t < tilesPerPool) ? W0 : W1;
    int tl = (t < tilesPerPool) ? t : t - tilesPerPool;
    int n  = tl * 16 + (lane & 15);
    int k  = c * 32 + (lane >> 4) * 16 + v * 2;
    dst[id] = f2bf_pk(W[(size_t)n * K + k], W[(size_t)n * K + k + 1]);
}

// ------------------- main fused attention-score GEMM ---------------------
// scores[p][b*510+s2] = sum_n w2p[n]*tanh( word[b,s2,:]·W1p[n,:] + b1p[n] )
__global__ __launch_bounds__(256) void k_scores(
    const float* __restrict__ lhs, const int* __restrict__ sep,
    const uint4* __restrict__ Bpack,
    const float* __restrict__ b1o, const float* __restrict__ w2o,
    const float* __restrict__ b1r, const float* __restrict__ w2r,
    float* __restrict__ scores) {
    __shared__ __align__(16) unsigned short As[32 * 776];  // 32 rows, padded
    __shared__ float scw[8][32];
    const int tid = threadIdx.x;
    const int m0  = blockIdx.x * 32;

    // ---- stage 32 gathered rows, fp32 -> bf16, into LDS ----
    for (int it = 0; it < 24; ++it) {
        int e  = it * 256 + tid;          // 0..6143 = 32 rows * 192 float4
        int g  = e / 192;
        int c4 = e - g * 192;
        int row = m0 + g;
        int b   = row / S2_;
        int s2  = row - b * S2_;
        int base = s2 + 1;
        int idx  = base + (base >= sep[b] ? 1 : 0);
        float4 src = ((const float4*)(lhs + ((size_t)(b * S_ + idx)) * D_))[c4];
        unsigned* p = (unsigned*)&As[g * 776 + c4 * 4];
        p[0] = f2bf_pk(src.x, src.y);
        p[1] = f2bf_pk(src.z, src.w);
    }
    __syncthreads();

    const int wid  = tid >> 5;
    const int lane = tid & 31;
    const int ln   = lane & 15;
    const int h    = lane >> 4;
    const int pool = wid >> 2;                    // waves 0-3: ori, 4-7: re
    const float* b1 = pool ? b1r : b1o;
    const float* w2 = pool ? w2r : w2o;
    const int t0 = wid * 6;                       // 6 N-tiles of 16 per wave

    v8f vzero = {0.f, 0.f, 0.f, 0.f, 0.f, 0.f, 0.f, 0.f};
    v8f acc[2][6];
#pragma unroll
    for (int m = 0; m < 2; ++m)
#pragma unroll
        for (int j = 0; j < 6; ++j) acc[m][j] = vzero;

    for (int c = 0; c < 24; ++c) {
        BF16x16 a[2];
#pragma unroll
        for (int m = 0; m < 2; ++m) {
            const uint4* p =
                (const uint4*)&As[(m * 16 + ln) * 776 + c * 32 + h * 16];
            a[m].u4[0] = p[0];        // K = c*32 + h*16 + 0..7
            a[m].u4[1] = p[1];        // K = c*32 + h*16 + 8..15
        }
        BF16x16 bf[6];
#pragma unroll
        for (int j = 0; j < 6; ++j) {
            const uint4* q = Bpack + (size_t)(((t0 + j) * 24 + c) * 32 + lane) * 2;
            bf[j].u4[0] = q[0];
            bf[j].u4[1] = q[1];
        }
        if (c + 1 < 24)
            __builtin_prefetch(
                (const void*)(Bpack + (size_t)((t0 * 24 + c + 1) * 32 + lane) * 2),
                0, 1);
#pragma unroll
        for (int m = 0; m < 2; ++m)
#pragma unroll
            for (int j = 0; j < 6; ++j)
                acc[m][j] = __builtin_amdgcn_wmma_f32_16x16x32_bf16(
                    false, a[m].v, false, bf[j].v, (short)0, acc[m][j],
                    false, false);
    }

    // ---- fused epilogue: +b1, tanh, *w2, sum over this wave's 96 cols ----
    float b1v[6], w2v[6];
#pragma unroll
    for (int j = 0; j < 6; ++j) {
        int n = (t0 + j - pool * 24) * 16 + ln;
        b1v[j] = b1[n];
        w2v[j] = w2[n];
    }
#pragma unroll
    for (int m = 0; m < 2; ++m) {
#pragma unroll
        for (int i = 0; i < 8; ++i) {
            float v = 0.f;
#pragma unroll
            for (int j = 0; j < 6; ++j)
                v += tanhf(acc[m][j][i] + b1v[j]) * w2v[j];
            v += __shfl_xor(v, 1, 32);
            v += __shfl_xor(v, 2, 32);
            v += __shfl_xor(v, 4, 32);
            v += __shfl_xor(v, 8, 32);
            if (ln == 0) scw[wid][m * 16 + h * 8 + i] = v;  // row-local
        }
    }
    __syncthreads();
    if (tid < 64) {   // deterministic fixed-order reduction across 4 waves
        int p = tid >> 5, r = tid & 31;
        float v = scw[p * 4 + 0][r] + scw[p * 4 + 1][r] +
                  scw[p * 4 + 2][r] + scw[p * 4 + 3][r];
        scores[(size_t)p * MROWS_ + m0 + r] = v;
    }
}

// -------------------- masked softmax over S2 per (pool,b) ----------------
__global__ __launch_bounds__(256) void k_softmax(
    const float* __restrict__ scores, const int* __restrict__ sep,
    const int* __restrict__ amask, float* __restrict__ alpha) {
    int b = blockIdx.x, p = blockIdx.y;
    __shared__ float red[8];
    __shared__ float s_max, s_sum;
    int sp = sep[b];
    const float* src = scores + (size_t)p * MROWS_ + b * S2_;
    float* dst = alpha + (size_t)p * MROWS_ + b * S2_;

    float xs[2];
    int   ms[2];
    float mx = -1e30f;
    for (int it = 0; it < 2; ++it) {
        int s = threadIdx.x + it * 256;
        float x = -1e30f;
        int m = 0;
        if (s < S2_) {
            int base = s + 1;
            int idx  = base + (base >= sp ? 1 : 0);
            m = amask[b * S_ + idx];
            x = m ? src[s] : -1e30f;
        }
        xs[it] = x; ms[it] = m;
        mx = fmaxf(mx, x);
    }
    for (int o = 16; o > 0; o >>= 1) mx = fmaxf(mx, __shfl_xor(mx, o, 32));
    if ((threadIdx.x & 31) == 0) red[threadIdx.x >> 5] = mx;
    __syncthreads();
    if (threadIdx.x == 0) {
        float m2 = red[0];
        for (int i = 1; i < 8; ++i) m2 = fmaxf(m2, red[i]);
        s_max = m2;
    }
    __syncthreads();
    mx = s_max;
    float es[2], sum = 0.f;
    for (int it = 0; it < 2; ++it) {
        float e = ms[it] ? __expf(xs[it] - mx) : 0.f;
        es[it] = e; sum += e;
    }
    for (int o = 16; o > 0; o >>= 1) sum += __shfl_xor(sum, o, 32);
    if ((threadIdx.x & 31) == 0) red[threadIdx.x >> 5] = sum;
    __syncthreads();
    if (threadIdx.x == 0) {
        float t = 0.f;
        for (int i = 0; i < 8; ++i) t += red[i];
        s_sum = t;
    }
    __syncthreads();
    float inv = 1.f / s_sum;
    for (int it = 0; it < 2; ++it) {
        int s = threadIdx.x + it * 256;
        if (s < S2_) dst[s] = es[it] * inv;
    }
}

// ------- alpha-weighted pooling (both pools, one streaming pass) ---------
__global__ __launch_bounds__(256) void k_pool(
    const float* __restrict__ lhs, const int* __restrict__ sep,
    const float* __restrict__ alpha, float* __restrict__ feat,
    unsigned short* __restrict__ featbf) {
    int b = blockIdx.x;
    __shared__ float ao_s[S2_], ar_s[S2_];
    for (int s = threadIdx.x; s < S2_; s += 256) {
        ao_s[s] = alpha[b * S2_ + s];
        ar_s[s] = alpha[MROWS_ + b * S2_ + s];
    }
    __syncthreads();
    int sp = sep[b];
    float ao[3] = {0.f, 0.f, 0.f}, ar[3] = {0.f, 0.f, 0.f};
    for (int s = 0; s < S2_; ++s) {
        int base = s + 1;
        int idx  = base + (base >= sp ? 1 : 0);
        const float* row = lhs + ((size_t)(b * S_ + idx)) * D_;
        float wo = ao_s[s], wr = ar_s[s];
#pragma unroll
        for (int u = 0; u < 3; ++u) {
            float x = row[threadIdx.x + u * 256];
            ao[u] += wo * x;
            ar[u] += wr * x;
        }
    }
#pragma unroll
    for (int u = 0; u < 3; ++u) {
        int c = threadIdx.x + u * 256;
        feat[(size_t)b * 1536 + c]        = ao[u];
        feat[(size_t)b * 1536 + 768 + c]  = ar[u];
        featbf[(size_t)b * 1536 + c]       = f2bf1(ao[u]);
        featbf[(size_t)b * 1536 + 768 + c] = f2bf1(ar[u]);
    }
}

// ------------------------ FC1 (WMMA) + ReLU ------------------------------
__global__ __launch_bounds__(256) void k_fc1(
    const unsigned short* __restrict__ featbf, const uint4* __restrict__ Wp,
    const float* __restrict__ bfc1, float* __restrict__ hbuf) {
    int wid = threadIdx.x >> 5, lane = threadIdx.x & 31;
    int ln = lane & 15, h = lane >> 4;
    int m0 = blockIdx.x * 16;                  // 8 M-tiles
    int t  = blockIdx.y * 8 + wid;             // 48 N-tiles
    v8f acc = {0.f, 0.f, 0.f, 0.f, 0.f, 0.f, 0.f, 0.f};
    for (int c = 0; c < 48; ++c) {
        BF16x16 a, bfr;
        const uint4* pa =
            (const uint4*)(featbf + (size_t)(m0 + ln) * 1536 + c * 32 + h * 16);
        a.u4[0] = pa[0];
        a.u4[1] = pa[1];
        const uint4* q = Wp + (size_t)((t * 48 + c) * 32 + lane) * 2;
        bfr.u4[0] = q[0];
        bfr.u4[1] = q[1];
        acc = __builtin_amdgcn_wmma_f32_16x16x32_bf16(
            false, a.v, false, bfr.v, (short)0, acc, false, false);
    }
#pragma unroll
    for (int i = 0; i < 8; ++i) {
        int row = m0 + h * 8 + i;
        int n   = t * 16 + ln;
        float v = acc[i] + bfc1[n];
        hbuf[(size_t)row * D_ + n] = v > 0.f ? v : 0.f;
    }
}

// ------------------------------ FC2 (tiny) -------------------------------
__global__ __launch_bounds__(192) void k_fc2(
    const float* __restrict__ hbuf, const float* __restrict__ Wfc2,
    const float* __restrict__ bfc2, float* __restrict__ out) {
    int b = blockIdx.x;
    int o = threadIdx.x >> 5, lane = threadIdx.x & 31;
    float s = 0.f;
    for (int k = lane; k < D_; k += 32)
        s += hbuf[(size_t)b * D_ + k] * Wfc2[o * D_ + k];
    for (int off = 16; off > 0; off >>= 1) s += __shfl_xor(s, off, 32);
    if (lane == 0) out[b * 6 + o] = s + bfc2[o];
}

extern "C" void kernel_launch(void* const* d_in, const int* in_sizes, int n_in,
                              void* d_out, int out_size, void* d_ws,
                              size_t ws_size, hipStream_t stream) {
    const float* lhs   = (const float*)d_in[0];
    const int*   ids   = (const int*)d_in[1];
    const int*   amask = (const int*)d_in[2];
    const float* W1o   = (const float*)d_in[3];
    const float* b1o   = (const float*)d_in[4];
    const float* w2o   = (const float*)d_in[5];
    const float* W1r   = (const float*)d_in[6];
    const float* b1r   = (const float*)d_in[7];
    const float* w2r   = (const float*)d_in[8];
    const float* Wfc1  = (const float*)d_in[9];
    const float* bfc1  = (const float*)d_in[10];
    const float* Wfc2  = (const float*)d_in[11];
    const float* bfc2  = (const float*)d_in[12];
    float* out = (float*)d_out;
    (void)in_sizes; (void)n_in; (void)out_size; (void)ws_size;

    char* ws = (char*)d_ws;
    size_t off = 0;
    auto take = [&](size_t bytes) -> char* {
        char* p = ws + off;
        off = (off + bytes + 255) & ~(size_t)255;
        return p;
    };
    int*      sep    = (int*)take((size_t)B_ * 4);
    unsigned* Bpack  = (unsigned*)take((size_t)48 * 24 * 32 * 8 * 4);  // 1.18MB
    unsigned* Wfc1p  = (unsigned*)take((size_t)48 * 48 * 32 * 8 * 4);  // 2.36MB
    float*    scores = (float*)take((size_t)2 * MROWS_ * 4);
    float*    alpha  = (float*)take((size_t)2 * MROWS_ * 4);
    float*    feat   = (float*)take((size_t)B_ * 1536 * 4);
    unsigned short* fbf = (unsigned short*)take((size_t)B_ * 1536 * 2);
    float*    hbuf   = (float*)take((size_t)B_ * D_ * 4);

    k_sep<<<B_, 256, 0, stream>>>(ids, sep);
    // W1o||W1r: 48 tiles, 24 K-chunks -> 294912 dwords
    k_pack_w<<<1152, 256, 0, stream>>>(W1o, W1r, Bpack, 24, D_, 24, 294912);
    // Wfc1: 48 tiles, 48 K-chunks -> 589824 dwords
    k_pack_w<<<2304, 256, 0, stream>>>(Wfc1, Wfc1, Wfc1p, 48, 1536, 48, 589824);
    k_scores<<<MROWS_ / 32, 256, 0, stream>>>(lhs, sep, (const uint4*)Bpack,
                                              b1o, w2o, b1r, w2r, scores);
    k_softmax<<<dim3(B_, 2), 256, 0, stream>>>(scores, sep, amask, alpha);
    k_pool<<<B_, 256, 0, stream>>>(lhs, sep, alpha, feat, fbf);
    k_fc1<<<dim3(8, 6), 256, 0, stream>>>(fbf, (const uint4*)Wfc1p, bfc1, hbuf);
    k_fc2<<<B_, 192, 0, stream>>>(hbuf, Wfc2, bfc2, out);
}
